// FFTConv_44573170598528
// MI455X (gfx1250) — compile-verified, matrix operand order checked by reading, not run on hardware
//
#include <hip/hip_runtime.h>

// ---------------------------------------------------------------------------
// Causal long convolution (Hyena FFTConv equivalent) + residual x*B
//   y[b,d,t] = sum_{tau<=t} h[d,t-tau] * x[b,d,tau] + x[b,d,t]*B[d]
// Block-Toeplitz GEMM on CDNA5 WMMA bf16 (16x16x32), hi/lo bf16 split for
// fp32-class accuracy, f32 accumulation. Row staging via Tensor Data Mover
// (tensor_load_to_lds, TENSORcnt), per ISA cdna5 ch.8 descriptor layout.
// ---------------------------------------------------------------------------

typedef __bf16 bf16_t;
typedef __attribute__((ext_vector_type(16))) __bf16 v16bf;
typedef __attribute__((ext_vector_type(8)))  __bf16 v8bf;
typedef __attribute__((ext_vector_type(8)))  float  v8f;
typedef __attribute__((ext_vector_type(4)))  float  v4f;
typedef __attribute__((ext_vector_type(4)))  unsigned int v4u;
typedef __attribute__((ext_vector_type(4)))  int    v4i;
typedef __attribute__((ext_vector_type(8)))  int    v8i;

#define D_CH    768
#define LSEQ    8192
#define NBATCH  4

// LDS layout (bytes):
//   [0      , 32768 ) : x row staged as f32 (TDM destination)
//   [32768  , 65536 ) : h row staged as f32 (TDM destination)
//   [65536  , ...   ) : bf16 arrays hh | hl | xh | xl
#define XS_F32_OFF 0
#define HS_F32_OFF 32768
#define BF_OFF     65536
#define HOFF 32
#define HSZ  (HOFF + LSEQ + 32)    // 8256 elements
#define XOFF 256
#define XSZ  (XOFF + LSEQ + 32)    // 8480 elements
#define SMEM_BYTES ((size_t)BF_OFF + (size_t)(2 * HSZ + 2 * XSZ) * sizeof(bf16_t)) // 132480

// Issue one TDM 2D load: contiguous row of 8192 f32 -> LDS byte offset.
// D# layout per CDNA5 ISA ch.8 (group0: count/lds_addr/global_addr/type=2;
// group1: data_size=4B, tensor_dim0=tile_dim0=8192, 1 row).
__device__ __forceinline__ void tdm_load_row_f32(unsigned lds_byte_off,
                                                 const float* gptr) {
    unsigned long long ga = (unsigned long long)(size_t)gptr;
    v4u g0;
    g0[0] = 1u;                                   // [1:0] count=1 (valid user D#)
    g0[1] = lds_byte_off;                         // [63:32] lds_addr (bytes)
    g0[2] = (unsigned)(ga & 0xFFFFFFFFu);         // [95:64] global_addr lo
    g0[3] = (unsigned)((ga >> 32) & 0x01FFFFFFu)  // [120:96] global_addr hi
          | (2u << 30);                           // [127:126] type=2 ("image")
    v8i g1;
    g1[0] = (int)(2u << 16);          // [17:16] data_size=2 -> 4 bytes; mask=0
    g1[1] = (int)(8192u << 16);       // [63:48] tensor_dim0[15:0] = 8192
    g1[2] = (int)(1u << 16);          // [79:64] dim0 hi=0 | [95:80] tensor_dim1=1
    g1[3] = (int)(8192u << 16);       // [111:96] dim1 hi=0 | [127:112] tile_dim0=8192
    g1[4] = 1;                        // [143:128] tile_dim1=1 | [159:144] tile_dim2=0
    g1[5] = 8192;                     // [191:160] tensor_dim0_stride lo = 8192
    g1[6] = (int)(8192u << 16);       // [207:192] stride0 hi=0 | [223:208] stride1 lo
    g1[7] = 0;                        // [255:224] stride1 hi
    v4i z4 = {0, 0, 0, 0};            // groups 2/3 unused (2D tensor)
    v8i z8 = {0, 0, 0, 0, 0, 0, 0, 0};
    // 6-arg form on this toolchain: (g0, g1, g2, g3, g4, cpol)
    __builtin_amdgcn_tensor_load_to_lds(g0, g1, z4, z4, z8, 0);
}

__global__ __launch_bounds__(128)
void fftconv_wmma_kernel(const float* __restrict__ h,
                         const float* __restrict__ x,
                         const float* __restrict__ Bg,
                         float* __restrict__ y)
{
    extern __shared__ __align__(16) unsigned char smem_raw[];
    float*  __restrict__ xs = (float*)(smem_raw + XS_F32_OFF);   // x row f32
    float*  __restrict__ hs = (float*)(smem_raw + HS_F32_OFF);   // h row f32
    bf16_t* __restrict__ hh = (bf16_t*)(smem_raw + BF_OFF);
    bf16_t* __restrict__ hl = hh + HSZ;
    bf16_t* __restrict__ xh = hl + HSZ;     // 16B aligned
    bf16_t* __restrict__ xl = xh + XSZ;

    const int tid = threadIdx.x;            // 128 threads = 4 waves
    const int d   = blockIdx.z;             // channel
    const int b   = blockIdx.y;             // batch

    const float* __restrict__ hrow = h + (size_t)d * LSEQ;
    const float* __restrict__ xrow = x + ((size_t)b * D_CH + d) * LSEQ;
    float*       __restrict__ yrow = y + ((size_t)b * D_CH + d) * LSEQ;

    // ---- wave 0: DMA both rows into LDS via Tensor Data Mover ----
    if (tid < 32) {
        tdm_load_row_f32(XS_F32_OFF, xrow);
        tdm_load_row_f32(HS_F32_OFF, hrow);
    }

    // ---- all threads: zero the bf16 pads (overlaps with TDM) ----
    for (int i = tid; i < HOFF; i += 128) { hh[i] = (bf16_t)0.0f; hl[i] = (bf16_t)0.0f; }
    for (int i = tid; i < 32;   i += 128) {
        hh[HOFF + LSEQ + i] = (bf16_t)0.0f; hl[HOFF + LSEQ + i] = (bf16_t)0.0f;
        xh[XOFF + LSEQ + i] = (bf16_t)0.0f; xl[XOFF + LSEQ + i] = (bf16_t)0.0f;
    }
    for (int i = tid; i < XOFF; i += 128) { xh[i] = (bf16_t)0.0f; xl[i] = (bf16_t)0.0f; }

    if (tid < 32) __builtin_amdgcn_s_wait_tensorcnt(0);
    __syncthreads();

    // ---- f32 (LDS) -> (hi, lo) bf16 split ----
    for (int t = tid; t < LSEQ; t += 128) {
        float hv = hs[t];
        bf16_t hhi = (bf16_t)hv;
        hh[HOFF + t] = hhi;
        hl[HOFF + t] = (bf16_t)(hv - (float)hhi);

        float xv = xs[t];
        bf16_t xhi = (bf16_t)xv;
        xh[XOFF + t] = xhi;
        xl[XOFF + t] = (bf16_t)(xv - (float)xhi);
    }
    __syncthreads();

    // ---- per-wave tile ----
    const int lane = tid & 31;
    const int w    = tid >> 5;
    const int g    = lane >> 4;      // K-half selector (0/1)
    const int mn   = lane & 15;      // A row m / B,C column n

    const int tile = blockIdx.x * 4 + w;   // 0..31 (16-chunk output tile)
    const int i0   = tile * 16;            // starting output chunk

    v8f acc = {};

    const int dmax = i0 + 16;
    for (int delta = 0; delta <= dmax; delta += 2) {
        // A fragments: h Toeplitz slice  A[m,k] = h[16*delta + m - k]
        v16bf a_hi, a_lo;
        const int abase = HOFF + 16 * delta + mn;
#pragma unroll
        for (int e = 0; e < 16; ++e) {
            const int K = (e & 7) + (g << 3) + ((e >> 3) << 4);
            a_hi[e] = hh[abase - K];
            a_lo[e] = hl[abase - K];
        }

        // B fragments: B[k,n] = x[16*(i0+n-delta) + k]; 16B-aligned K-runs
        const int xb = XOFF + 16 * (i0 + mn - delta) + (g << 3);
        v8bf b0h = *(const v8bf*)(xh + xb);
        v8bf b1h = *(const v8bf*)(xh + xb + 16);
        v8bf b0l = *(const v8bf*)(xl + xb);
        v8bf b1l = *(const v8bf*)(xl + xb + 16);
        v16bf b_hi = __builtin_shufflevector(b0h, b1h,
                         0, 1, 2, 3, 4, 5, 6, 7, 8, 9, 10, 11, 12, 13, 14, 15);
        v16bf b_lo = __builtin_shufflevector(b0l, b1l,
                         0, 1, 2, 3, 4, 5, 6, 7, 8, 9, 10, 11, 12, 13, 14, 15);

        // split-precision product: hi*hi + hi*lo + lo*hi (f32 accumulate)
        acc = __builtin_amdgcn_wmma_f32_16x16x32_bf16(false, a_hi, false, b_hi,
                                                      (short)0, acc, false, false);
        acc = __builtin_amdgcn_wmma_f32_16x16x32_bf16(false, a_hi, false, b_lo,
                                                      (short)0, acc, false, false);
        acc = __builtin_amdgcn_wmma_f32_16x16x32_bf16(false, a_lo, false, b_hi,
                                                      (short)0, acc, false, false);
    }

    // ---- epilogue: residual (x from LDS staging) + vector stores ----
    // C layout: lanes 0-15 -> N=lane, M=v ; lanes 16-31 -> N=lane-16, M=8+v
    const float Bd = Bg[d];
    const int t0 = 16 * (i0 + mn) + (g << 3);   // 16B aligned
    v4f o0, o1;
#pragma unroll
    for (int v = 0; v < 4; ++v) {
        o0[v] = acc[v]     + xs[t0 + v]     * Bd;
        o1[v] = acc[4 + v] + xs[t0 + 4 + v] * Bd;
    }
    *(v4f*)(yrow + t0)     = o0;
    *(v4f*)(yrow + t0 + 4) = o1;
}

extern "C" void kernel_launch(void* const* d_in, const int* in_sizes, int n_in,
                              void* d_out, int out_size, void* d_ws, size_t ws_size,
                              hipStream_t stream) {
    (void)in_sizes; (void)n_in; (void)out_size; (void)d_ws; (void)ws_size;

    const float* h  = (const float*)d_in[0];   // [768, 8192]
    const float* x  = (const float*)d_in[1];   // [4, 768, 8192]
    const float* Bp = (const float*)d_in[2];   // [768, 1]
    float*       y  = (float*)d_out;           // [4, 768, 8192]

    // allow >64KB dynamic LDS (gfx1250: up to 320KB per workgroup)
    (void)hipFuncSetAttribute((const void*)fftconv_wmma_kernel,
                              hipFuncAttributeMaxDynamicSharedMemorySize,
                              (int)SMEM_BYTES);

    dim3 grid(8 /*i-tile groups*/, NBATCH, D_CH);
    dim3 block(128);
    fftconv_wmma_kernel<<<grid, block, SMEM_BYTES, stream>>>(h, x, Bp, y);
}
